// SelfAttention_39788577030790
// MI455X (gfx1250) — matrix operand order repaired and checked by reading, not compiled, not run on hardware
//
#include <hip/hip_runtime.h>
#include <hip/hip_bf16.h>
#include <math.h>

// Problem constants (fixed by the reference)
#define BATCH 8
#define CCH   256     // channels C
#define NTOK  4096    // H*W
#define DH    32      // d = C/8

typedef __attribute__((ext_vector_type(16))) __bf16 v16bf;
typedef __attribute__((ext_vector_type(8)))  __bf16 v8bf;
typedef __attribute__((ext_vector_type(8)))  float  v8f;

__device__ __forceinline__ v8bf ld8(const __bf16* p) {
    return *reinterpret_cast<const v8bf*>(p);
}
__device__ __forceinline__ v16bf cat8(v8bf lo, v8bf hi) {
    return __builtin_shufflevector(lo, hi, 0,1,2,3,4,5,6,7,8,9,10,11,12,13,14,15);
}
__device__ __forceinline__ v8f wmma_bf16(v16bf a, v16bf b, v8f c) {
    // D = A(16x32 bf16) * B(32x16 bf16) + C(16x16 f32)
    return __builtin_amdgcn_wmma_f32_16x16x32_bf16(
        /*neg_a=*/false, a, /*neg_b=*/false, b,
        /*c_mod=*/(short)0, c, /*reuse_a=*/false, /*reuse_b=*/false);
}

// ---------------------------------------------------------------------------
// Kernel 1: fused QKV projection (K=256, ~5.4 GFLOP -> VALU fp32 is fine).
//   q -> Qb[b][n][d] bf16 (WMMA fragment row layout)
//   k -> Kb[b][n][d] bf16
//   v -> Vb[b][c][n] bf16 (m-contiguous rows: stage-3 B fragments)
// One thread owns one token n; 10 passes of 32 output rows, W staged in LDS.
// ---------------------------------------------------------------------------
__global__ __launch_bounds__(256) void proj_kernel(
    const float* __restrict__ x,
    const float* __restrict__ Wq, const float* __restrict__ bq,
    const float* __restrict__ Wk, const float* __restrict__ bk,
    const float* __restrict__ Wv, const float* __restrict__ bv,
    __bf16* __restrict__ Qb, __bf16* __restrict__ Kb, __bf16* __restrict__ Vb)
{
    __shared__ float Wl[CCH * 32];            // [c][r] 32 KB
    const int tid = threadIdx.x;
    const int n   = blockIdx.x * 256 + tid;   // token
    const int b   = blockIdx.z;
    const int rowBase = blockIdx.y * 32;      // 10 passes x 32 rows = 320

    for (int i = tid; i < CCH * 32; i += 256) {
        const int c = i >> 5, r = i & 31;
        const int row = rowBase + r;
        float wv;
        if (row < 32)       wv = Wq[row * CCH + c];
        else if (row < 64)  wv = Wk[(row - 32) * CCH + c];
        else                wv = Wv[(row - 64) * CCH + c];
        Wl[c * 32 + r] = wv;
    }
    __syncthreads();

    float acc[32];
#pragma unroll
    for (int r = 0; r < 32; ++r) acc[r] = 0.f;

    const float* xb = x + ((size_t)b * CCH) * NTOK + n;
    for (int c = 0; c < CCH; ++c) {
        const float xv = xb[(size_t)c * NTOK];
        const float4* w4 = reinterpret_cast<const float4*>(&Wl[c * 32]);
#pragma unroll
        for (int q = 0; q < 8; ++q) {
            const float4 w = w4[q];
            acc[q * 4 + 0] = fmaf(w.x, xv, acc[q * 4 + 0]);
            acc[q * 4 + 1] = fmaf(w.y, xv, acc[q * 4 + 1]);
            acc[q * 4 + 2] = fmaf(w.z, xv, acc[q * 4 + 2]);
            acc[q * 4 + 3] = fmaf(w.w, xv, acc[q * 4 + 3]);
        }
    }

#pragma unroll
    for (int r = 0; r < 32; ++r) {
        const int row = rowBase + r;              // uniform per pass
        float bias;
        if (row < 32)      bias = bq[row];
        else if (row < 64) bias = bk[row - 32];
        else               bias = bv[row - 64];
        const float val = acc[r] + bias;
        if (row < 32) {
            Qb[((size_t)b * NTOK + n) * DH + row] = (__bf16)val;
        } else if (row < 64) {
            Kb[((size_t)b * NTOK + n) * DH + (row - 32)] = (__bf16)val;
        } else {
            Vb[((size_t)b * CCH + (row - 64)) * NTOK + n] = (__bf16)val;
        }
    }
}

// ---------------------------------------------------------------------------
// Kernel 2: per-column softmax stats over the query axis n:
//   M[m] = max_n e[n,m],  Z[m] = sum_n exp(e[n,m]-M[m]);  store M and 1/Z.
// Wave owns 16 columns (K B-fragment resident in VGPRs for the whole loop).
// Two 16-row WMMA tiles per online update to amortize the rescale/bpermute.
// ---------------------------------------------------------------------------
__global__ __launch_bounds__(256) void stats_kernel(
    const __bf16* __restrict__ Qb, const __bf16* __restrict__ Kb,
    float* __restrict__ Mb, float* __restrict__ Rb)
{
    const int tid  = threadIdx.x;
    const int w    = tid >> 5, lane = tid & 31;
    const int l15  = lane & 15, half = lane >> 4;
    const int b    = blockIdx.y;
    const int mBase = blockIdx.x * 128 + w * 16;

    const __bf16* krow = Kb + ((size_t)b * NTOK + mBase + l15) * DH + half * 8;
    const v16bf kfrag = cat8(ld8(krow), ld8(krow + 16));

    float runM = -INFINITY, runS = 0.f;
    const v8f cz = {0.f, 0.f, 0.f, 0.f, 0.f, 0.f, 0.f, 0.f};

    for (int n0 = 0; n0 < NTOK; n0 += 32) {
        const __bf16* qrow0 = Qb + ((size_t)b * NTOK + n0 + l15) * DH;
        const __bf16* qrow1 = qrow0 + 16 * DH;
        const v16bf qa0 = cat8(ld8(qrow0 + half * 8), ld8(qrow0 + 16 + half * 8));
        const v16bf qa1 = cat8(ld8(qrow1 + half * 8), ld8(qrow1 + 16 + half * 8));
        const v8f e0 = wmma_bf16(qa0, kfrag, cz);
        const v8f e1 = wmma_bf16(qa1, kfrag, cz);

        float tmax = fmaxf(e0[0], e1[0]);
#pragma unroll
        for (int j = 1; j < 8; ++j) tmax = fmaxf(tmax, fmaxf(e0[j], e1[j]));
        tmax = fmaxf(tmax, __shfl_xor(tmax, 16, 32));   // combine row halves
        const float newM = fmaxf(runM, tmax);

        float s = 0.f;
#pragma unroll
        for (int j = 0; j < 8; ++j)
            s += __expf(e0[j] - newM) + __expf(e1[j] - newM);
        s += __shfl_xor(s, 16, 32);

        runS = runS * __expf(runM - newM) + s;          // online rescale
        runM = newM;
    }

    if (lane < 16) {
        const size_t idx = (size_t)b * NTOK + mBase + lane;
        Mb[idx] = runM;
        Rb[idx] = 1.0f / runS;
    }
}

// ---------------------------------------------------------------------------
// Kernel 3: out^T[n,c] = sum_m P[n,m]*V[m,c],  P = exp(e-M[m]) * (1/Z[m]).
// Workgroup: 32 n x 256 c output tile; m-loop in steps of 64.
//   phase1: ALL 8 waves recompute one 16x16 E tile (32n x 64m total) by WMMA,
//           apply exp*(1/Z), stage P (bf16) in a 32x64 LDS tile (layout
//           conversion C-fragment -> A-fragment).
//   phase2: each wave: 2 P A-fragments x 4 V B-fragments = 8 WMMAs into its
//           16n x 64c accumulators.
// Per-batch Vb slice (2 MB) is L2-resident across the 64 workgroups of the
// batch -> stage stays v_wmma-bound; prefetch the next V rows each iteration.
// ---------------------------------------------------------------------------
__global__ __launch_bounds__(256) void attn_kernel(
    const __bf16* __restrict__ Qb, const __bf16* __restrict__ Kb,
    const __bf16* __restrict__ Vb,
    const float* __restrict__ Mb, const float* __restrict__ Rb,
    float* __restrict__ out)
{
    __shared__ __bf16 Pl[32 * 64];            // P staging tile, 4 KB
    const int tid  = threadIdx.x;
    const int w    = tid >> 5, lane = tid & 31;
    const int l15  = lane & 15, half = lane >> 4;
    const int b    = blockIdx.y;
    const int nBase = blockIdx.x * 32;
    const int wn = w >> 2, wc = w & 3;        // wave tile: 16 n x 64 c

    const v8f cz = {0.f, 0.f, 0.f, 0.f, 0.f, 0.f, 0.f, 0.f};
    v8f acc[4] = {cz, cz, cz, cz};

    // loop-invariant Q A-fragment (rows nBase + wn*16) — same rows as phase 2
    const __bf16* qrow = Qb + ((size_t)b * NTOK + nBase + wn * 16 + l15) * DH;
    const v16bf qa = cat8(ld8(qrow + half * 8), ld8(qrow + 16 + half * 8));

    const float* Mrow = Mb + (size_t)b * NTOK;
    const float* Rrow = Rb + (size_t)b * NTOK;

    for (int m0 = 0; m0 < NTOK; m0 += 64) {
        // ---- phase 1: E tile (16n x 16m per wave), P -> LDS -------------
        {
            const int mcol = m0 + wc * 16;    // this wave's 16 m columns
            const __bf16* krow =
                Kb + ((size_t)b * NTOK + mcol + l15) * DH + half * 8;
            const v16bf kfrag = cat8(ld8(krow), ld8(krow + 16));
            const v8f e = wmma_bf16(qa, kfrag, cz);

            const int mg = mcol + l15;        // this lane's column
            const float Mm = Mrow[mg];
            const float rz = Rrow[mg];
            const int colL = wc * 16 + l15;
            const int rowB = wn * 16 + half * 8;
#pragma unroll
            for (int j = 0; j < 8; ++j)
                Pl[(rowB + j) * 64 + colL] = (__bf16)(__expf(e[j] - Mm) * rz);
        }

        // prefetch next iteration's V rows while P settles
        if (m0 + 64 < NTOK) {
#pragma unroll
            for (int j = 0; j < 4; ++j) {
                const int c = wc * 64 + j * 16 + l15;
                __builtin_prefetch(
                    Vb + ((size_t)b * CCH + c) * NTOK + (m0 + 64) + half * 16,
                    0, 3);
            }
        }
        __syncthreads();

        // ---- phase 2: P (A frags from LDS) x V (B frags from global) ----
        const __bf16* prow = &Pl[(wn * 16 + l15) * 64];
        const v16bf pa0 = cat8(*reinterpret_cast<const v8bf*>(prow + half * 8),
                               *reinterpret_cast<const v8bf*>(prow + 16 + half * 8));
        const v16bf pa1 = cat8(*reinterpret_cast<const v8bf*>(prow + 32 + half * 8),
                               *reinterpret_cast<const v8bf*>(prow + 48 + half * 8));
#pragma unroll
        for (int j = 0; j < 4; ++j) {
            const int c = wc * 64 + j * 16 + l15;
            const __bf16* vrow = Vb + ((size_t)b * CCH + c) * NTOK + m0 + half * 8;
            const v16bf vf0 = cat8(ld8(vrow), ld8(vrow + 16));
            const v16bf vf1 = cat8(ld8(vrow + 32), ld8(vrow + 48));
            acc[j] = wmma_bf16(pa0, vf0, acc[j]);
            acc[j] = wmma_bf16(pa1, vf1, acc[j]);
        }
        __syncthreads();
    }

    // store: C-fragment rows are 8 consecutive n for fixed c -> 2x b128 each
#pragma unroll
    for (int j = 0; j < 4; ++j) {
        const int c = wc * 64 + j * 16 + l15;
        const size_t base =
            ((size_t)b * CCH + c) * NTOK + nBase + wn * 16 + half * 8;
        float4 lo, hi;
        lo.x = acc[j][0]; lo.y = acc[j][1]; lo.z = acc[j][2]; lo.w = acc[j][3];
        hi.x = acc[j][4]; hi.y = acc[j][5]; hi.z = acc[j][6]; hi.w = acc[j][7];
        *reinterpret_cast<float4*>(out + base)     = lo;
        *reinterpret_cast<float4*>(out + base + 4) = hi;
    }
}

// ---------------------------------------------------------------------------
extern "C" void kernel_launch(void* const* d_in, const int* in_sizes, int n_in,
                              void* d_out, int out_size, void* d_ws, size_t ws_size,
                              hipStream_t stream)
{
    const float* x  = (const float*)d_in[0];
    const float* Wq = (const float*)d_in[1];
    const float* bq = (const float*)d_in[2];
    const float* Wk = (const float*)d_in[3];
    const float* bk = (const float*)d_in[4];
    const float* Wv = (const float*)d_in[5];
    const float* bv = (const float*)d_in[6];
    float* out = (float*)d_out;

    // Workspace layout:
    //   Qb bf16 [B][N][32] : 2 MiB     Kb bf16 [B][N][32] : 2 MiB
    //   Vb bf16 [B][C][N]  : 16 MiB    Mb/Rb f32 [B][N]   : 128 KiB each
    char* ws = (char*)d_ws;
    __bf16* Qb = (__bf16*)(ws);
    __bf16* Kb = (__bf16*)(ws + (2ull << 20));
    __bf16* Vb = (__bf16*)(ws + (4ull << 20));
    float*  Mb = (float*)(ws + (20ull << 20));
    float*  Rb = (float*)(ws + (20ull << 20) + (BATCH * NTOK * sizeof(float)));

    // 1) fused QKV projection
    proj_kernel<<<dim3(NTOK / 256, 10, BATCH), 256, 0, stream>>>(
        x, Wq, bq, Wk, bk, Wv, bv, Qb, Kb, Vb);

    // 2) per-column softmax statistics (max + 1/sum)
    stats_kernel<<<dim3(NTOK / 128, BATCH), 256, 0, stream>>>(Qb, Kb, Mb, Rb);

    // 3) fused E-recompute + softmax-apply + output GEMM
    attn_kernel<<<dim3(NTOK / 64, BATCH), 256, 0, stream>>>(
        Qb, Kb, Vb, Mb, Rb, out);
}
